// WindowedSelfAttention3D_7267084665092
// MI455X (gfx1250) — matrix operand order repaired and checked by reading, not compiled, not run on hardware
//
#include <hip/hip_runtime.h>

// ---------------------------------------------------------------------------
// Windowed 3D self-attention for MI455X (gfx1250), bf16 WMMA + f32 accumulate.
// One workgroup (8 wave32) per 4x4x4 window; all intermediates in LDS.
// X window is staged by the Tensor Data Mover (async DMA) while waves stage
// the weight matrices; QKV / scores / PV / projection run on v_wmma bf16.
// ---------------------------------------------------------------------------

#define C_DIM   64
#define HEADS   4
#define HD      16            // head dim
#define DHW     (64*64*64)    // 262144
#define LDB     72            // padded row stride (elements) for bf16 matrices
#define LDSROW  72            // padded row stride (f32) for score matrix

typedef __attribute__((ext_vector_type(16))) __bf16 v16bf;
typedef __attribute__((ext_vector_type(2)))  __bf16 bf16x2;
typedef __attribute__((ext_vector_type(8)))  float  v8f;
typedef __attribute__((ext_vector_type(4)))  unsigned int u32x4;
typedef __attribute__((ext_vector_type(8)))  int    i32x8;
typedef __attribute__((ext_vector_type(4)))  int    i32x4;

#if __has_builtin(__builtin_amdgcn_tensor_load_to_lds)
#define USE_TDM 1
#if __has_include(<hip/amd_detail/amd_gfx1250_TDM.h>)
#define TDM_6ARG 1      // therock-10.0 toolchain: 6-arg builtin
#endif
#endif

union FragBF { uint4 u[2]; v16bf v; };
static_assert(sizeof(FragBF) == 32, "frag size");

__device__ __forceinline__ unsigned short f2bf(float f) {
  return __builtin_bit_cast(unsigned short, (__bf16)f);
}
// one v_cvt_pk_bf16_f32
__device__ __forceinline__ unsigned pk2(float a, float b) {
  bf16x2 v;
  v.x = (__bf16)a;
  v.y = (__bf16)b;
  return __builtin_bit_cast(unsigned, v);
}

// A-matrix 16x32 bf16 fragment (M x K), K contiguous in memory, row stride ldk.
// Layout (ISA 7.12.2): lanes 0-15 m=lane, K=khalf..khalf+7 in v0-3, +16 in v4-7.
__device__ __forceinline__ v16bf loadA32(const unsigned short* base, int row0,
                                         int kb, int ldk, int lane) {
  const int m     = lane & 15;
  const int khalf = (lane >> 4) << 3;     // 0 or 8
  const unsigned short* p = base + (row0 + m) * ldk + kb + khalf;
  FragBF f;
  f.u[0] = *(const uint4*)(p);            // K = kb+khalf .. +7
  f.u[1] = *(const uint4*)(p + 16);       // K = kb+16+khalf .. +7
  return f.v;
}

// A fragment with K=16 zero-padded to 32 (attention scores, head dim 16).
__device__ __forceinline__ v16bf loadA16(const unsigned short* base, int row0,
                                         int ldk, int lane) {
  const int m     = lane & 15;
  const int khalf = (lane >> 4) << 3;
  FragBF f;
  f.u[0] = *(const uint4*)(base + (row0 + m) * ldk + khalf);
  f.u[1] = make_uint4(0u, 0u, 0u, 0u);
  return f.v;
}

// B-matrix 32x16 bf16 fragment (K x N) read from a transposed store BT[n][k]
// (K contiguous, row stride ldk). n=lane&15, K=kb+koff..+7 in v0-3, +16 in v4-7
// (koff = 0 or 8), matching loadA32.
__device__ __forceinline__ v16bf loadB32(const unsigned short* base, int n0,
                                         int kb, int ldk, int lane) {
  const int n    = lane & 15;
  const int koff = (lane >> 4) << 3;      // 0 or 8
  const unsigned short* p = base + (n0 + n) * ldk + kb + koff;
  FragBF f;
  f.u[0] = *(const uint4*)(p);            // K = +0..7
  f.u[1] = *(const uint4*)(p + 16);       // K = +16..23
  return f.v;
}

// B fragment with K=16 zero-padded to 32 (v0-3: K=khalf..+7, v4-7: zero).
__device__ __forceinline__ v16bf loadB16(const unsigned short* base, int n0,
                                         int ldk, int lane) {
  const int n = lane & 15;
  const int khalf = (lane >> 4) << 3;
  const unsigned short* p = base + (n0 + n) * ldk + khalf;
  FragBF f;
  f.u[0] = *(const uint4*)(p);
  f.u[1] = make_uint4(0u, 0u, 0u, 0u);
  return f.v;
}

#define WMMA_BF16(a, b, c) \
  __builtin_amdgcn_wmma_f32_16x16x32_bf16(false, (a), false, (b), (short)0, (c), false, false)

// ---- LDS pool layout (bytes) ----
#define OFF_W     0                      // Wq|Wk|Wv|Wp bf16 [4][64][LDB]  36864
#define OFF_BIAS  36864                  // biases f32 [4][64]              1024
#define OFF_XT    37888                  // Xt/Ot bf16 [64][LDB] (aliased)  9216
#define OFF_QT    47104                  // Qt bf16 [l][c] (pre-scaled)     9216
#define OFF_KT    56320                  // Kt bf16 [l][c]                  9216
#define OFF_VD    65536                  // Vd bf16 [c][l]                  9216
#define OFF_S     74752                  // scores f32 [4][64][LDSROW]     73728
#define OFF_STAGE OFF_S                  // TDM staging f32 [c][l] 16384 (alias)
#define SMEM_BYTES 148480

__global__ __launch_bounds__(256)
void win_attn3d_kernel(const float* __restrict__ x,
                       const float* __restrict__ Wq, const float* __restrict__ bq,
                       const float* __restrict__ Wk, const float* __restrict__ bk,
                       const float* __restrict__ Wv, const float* __restrict__ bv,
                       const float* __restrict__ Wp, const float* __restrict__ bp,
                       float* __restrict__ out)
{
  __shared__ __align__(16) unsigned char smem[SMEM_BYTES];
  unsigned short* Wbf  = (unsigned short*)(smem + OFF_W);
  float*          bias = (float*)(smem + OFF_BIAS);
  unsigned short* Xt   = (unsigned short*)(smem + OFF_XT);   // [token][chan]
  unsigned short* Ot   = Xt;                                  // alias, [token][chan]
  unsigned short* Qt   = (unsigned short*)(smem + OFF_QT);   // [token][chan], *scale
  unsigned short* Kt   = (unsigned short*)(smem + OFF_KT);   // [token][chan]
  unsigned short* Vd   = (unsigned short*)(smem + OFF_VD);   // [chan][token]
  float*          S    = (float*)(smem + OFF_S);             // [head][i][LDSROW]

  const int t    = threadIdx.x;
  const int lane = t & 31;
  const int wv   = t >> 5;

  // ---- window decode: win = ((b*16 + wd)*16 + wh)*16 + ww ----
  const int win = blockIdx.x;
  const int b   = win >> 12;
  const int d0  = ((win >> 8) & 15) << 2;
  const int h0  = ((win >> 4) & 15) << 2;
  const int w0  = (win & 15) << 2;

  // ====== Phase 0a: kick off TDM load of the X window tile (wave 0 only) ======
#ifdef USE_TDM
  if (wv == 0) {
    // 4-D tile: (w=4, h=4, d=4, c=64); strides 64 / 4096 / 262144 f32 elements.
    const unsigned long long ga =
        (unsigned long long)(const void*)(x + (size_t)b * 64 * DHW +
                                          (size_t)d0 * 4096 + h0 * 64 + w0);
    u32x4 g0;
    g0[0] = 1u;                                           // count=1, user mode
    g0[1] = (unsigned)OFF_STAGE;                          // lds_addr (bytes)
    g0[2] = (unsigned)(ga & 0xFFFFFFFFu);                 // global_addr[31:0]
    g0[3] = (unsigned)((ga >> 32) & 0x01FFFFFFu) | (2u << 30);  // [56:32], type=2
    i32x8 g1;
    g1[0] = 0x00020000;                // data_size=4B; no multicast/pad/iterate
    g1[1] = (int)(64u << 16);          // tensor_dim0 = 64 (bits 79:48 lo)
    g1[2] = (int)(64u << 16);          // tensor_dim1 = 64
    g1[3] = (int)(4u << 16);           // tile_dim0 = 4 (w)
    g1[4] = (int)(4u | (4u << 16));    // tile_dim1 = 4 (h), tile_dim2 = 4 (d)
    g1[5] = 64;                        // tensor_dim0_stride = 64 (h line)
    g1[6] = (int)(4096u << 16);        // tensor_dim1_stride = 4096 (d plane)
    g1[7] = 0;
    i32x4 g2;
    g2[0] = 64;                        // tensor_dim2 = 64
    g2[1] = 64;                        // tensor_dim3 = 64
    g2[2] = 262144;                    // tensor_dim2_stride = DHW (channel)
    g2[3] = (int)(64u << 16);          // tile_dim3 = 64 (c)
    i32x4 g3;
    g3[0] = 0; g3[1] = 0; g3[2] = 0; g3[3] = 0;
#ifdef TDM_6ARG
    i32x8 g4;
    g4[0] = 0; g4[1] = 0; g4[2] = 0; g4[3] = 0;
    g4[4] = 0; g4[5] = 0; g4[6] = 0; g4[7] = 0;
    __builtin_amdgcn_tensor_load_to_lds(g0, g1, g2, g3, g4, 0);
#else
    __builtin_amdgcn_tensor_load_to_lds(g0, g1, g2, g3, 0);
#endif
  }
#endif

  // ========== Phase 0b: stage weights + biases (overlaps the TDM DMA) =========
#pragma unroll
  for (int i = 0; i < 16; ++i) {                      // 4096 float4 rows of W
    const int idx4 = t + 256 * i;
    const int g  = idx4 >> 10;
    const int o  = (idx4 >> 4) & 63;
    const int iq = (idx4 & 15) << 2;
    const float* W = (g == 0) ? Wq : (g == 1) ? Wk : (g == 2) ? Wv : Wp;
    const float4 w4 = *(const float4*)(W + o * 64 + iq);
    *(uint2*)(Wbf + g * (64 * LDB) + o * LDB + iq) =
        make_uint2(pk2(w4.x, w4.y), pk2(w4.z, w4.w));
  }
  {
    const int g = t >> 6, c = t & 63;
    const float* bg = (g == 0) ? bq : (g == 1) ? bk : (g == 2) ? bv : bp;
    bias[t] = bg[c];
  }

#ifdef USE_TDM
  if (wv == 0) __builtin_amdgcn_s_wait_tensorcnt(0);   // DMA done before barrier
  __syncthreads();
  // ---- convert staged f32 [c][l] -> bf16 Xt[l][c] ----
  const float* stage = (const float*)(smem + OFF_STAGE);
#pragma unroll
  for (int i = 0; i < 4; ++i) {
    const int rr = t + 256 * i;                       // 1024 float4 groups
    const int c  = rr >> 4;
    const int lq = (rr & 15) << 2;
    const float4 xv = *(const float4*)(stage + c * 64 + lq);
    unsigned short* dst = Xt + c;
    dst[(lq + 0) * LDB] = f2bf(xv.x);
    dst[(lq + 1) * LDB] = f2bf(xv.y);
    dst[(lq + 2) * LDB] = f2bf(xv.z);
    dst[(lq + 3) * LDB] = f2bf(xv.w);
  }
#else
  // ---- fallback: per-lane global gather of the window ----
#pragma unroll
  for (int i = 0; i < 4; ++i) {
    const int rr = t + 256 * i;                       // (c, dz, hy)
    const int c  = rr >> 4;
    const int dz = (rr >> 2) & 3;
    const int hy = rr & 3;
    const size_t off = (size_t)(b * 64 + c) * DHW +
                       (size_t)(d0 + dz) * 4096 + (h0 + hy) * 64 + w0;
    const float4 xv = *(const float4*)(x + off);
    const int l0 = dz * 16 + hy * 4;
    unsigned short* dst = Xt + c;
    dst[(l0 + 0) * LDB] = f2bf(xv.x);
    dst[(l0 + 1) * LDB] = f2bf(xv.y);
    dst[(l0 + 2) * LDB] = f2bf(xv.z);
    dst[(l0 + 3) * LDB] = f2bf(xv.w);
  }
#endif
  __syncthreads();

  // ============ Phase 1: Q/K/V = W @ X + b (3 GEMMs x 2 tiles/wave) ===========
  const float scale = 0.25f;                           // 16^-0.5
  const int N  = lane & 15;
  const int mh = (lane >> 4) << 3;
#pragma unroll
  for (int g = 0; g < 3; ++g) {                        // compile-time constant
#pragma unroll
    for (int j = 0; j < 2; ++j) {
      const int tile = wv * 2 + j;                     // 0..15
      const int m0 = (tile >> 2) << 4;
      const int n0 = (tile & 3) << 4;
      v8f acc = {};
#pragma unroll
      for (int kb = 0; kb < 64; kb += 32) {
        v16bf a  = loadA32(Wbf + g * (64 * LDB), m0, kb, LDB, lane);
        v16bf bb = loadB32(Xt, n0, kb, LDB, lane);
        acc = WMMA_BF16(a, bb, acc);
      }
      const float4 b0 = *(const float4*)(bias + g * 64 + m0 + mh);
      const float4 b1 = *(const float4*)(bias + g * 64 + m0 + mh + 4);
      const float v0 = acc[0] + b0.x, v1 = acc[1] + b0.y;
      const float v2 = acc[2] + b0.z, v3 = acc[3] + b0.w;
      const float v4 = acc[4] + b1.x, v5 = acc[5] + b1.y;
      const float v6 = acc[6] + b1.z, v7 = acc[7] + b1.w;
      const int l = n0 + N;
      if (g == 0) {                                    // Qt[l][c], pre-scaled
        *(uint2*)(Qt + l * LDB + m0 + mh) =
            make_uint2(pk2(v0 * scale, v1 * scale), pk2(v2 * scale, v3 * scale));
        *(uint2*)(Qt + l * LDB + m0 + mh + 4) =
            make_uint2(pk2(v4 * scale, v5 * scale), pk2(v6 * scale, v7 * scale));
      } else if (g == 1) {                             // Kt[l][c]
        *(uint2*)(Kt + l * LDB + m0 + mh) =
            make_uint2(pk2(v0, v1), pk2(v2, v3));
        *(uint2*)(Kt + l * LDB + m0 + mh + 4) =
            make_uint2(pk2(v4, v5), pk2(v6, v7));
      } else {                                         // Vd[c][l] (strided)
        Vd[(m0 + mh + 0) * LDB + l] = f2bf(v0);
        Vd[(m0 + mh + 1) * LDB + l] = f2bf(v1);
        Vd[(m0 + mh + 2) * LDB + l] = f2bf(v2);
        Vd[(m0 + mh + 3) * LDB + l] = f2bf(v3);
        Vd[(m0 + mh + 4) * LDB + l] = f2bf(v4);
        Vd[(m0 + mh + 5) * LDB + l] = f2bf(v5);
        Vd[(m0 + mh + 6) * LDB + l] = f2bf(v6);
        Vd[(m0 + mh + 7) * LDB + l] = f2bf(v7);
      }
    }
  }
  __syncthreads();

  // ========== Phase 2: S = (Q*scale)^T K per head (64 tiles, 8/wave) ==========
#pragma unroll
  for (int j = 0; j < 8; ++j) {
    const int tile = wv * 8 + j;                       // 0..63
    const int h  = tile >> 4;
    const int i0 = ((tile >> 2) & 3) << 4;
    const int j0 = (tile & 3) << 4;
    v16bf a  = loadA16(Qt + h * HD, i0, LDB, lane);    // K=16 zero-padded
    v16bf bb = loadB16(Kt + h * HD, j0, LDB, lane);
    v8f acc = {};
    acc = WMMA_BF16(a, bb, acc);
    float* dst = S + h * (64 * LDSROW);
#pragma unroll
    for (int r = 0; r < 8; ++r)
      dst[(i0 + r + mh) * LDSROW + j0 + N] = acc[r];
  }
  __syncthreads();

  // ================= Phase 3: softmax, one row per thread =====================
  {
    const int h = t >> 6, i = t & 63;
    float* row = S + h * (64 * LDSROW) + i * LDSROW;
    float mx = -3.0e38f;
#pragma unroll
    for (int j2 = 0; j2 < 64; j2 += 4) {
      float4 v4 = *(float4*)(row + j2);
      mx = fmaxf(mx, fmaxf(fmaxf(v4.x, v4.y), fmaxf(v4.z, v4.w)));
    }
    float sum = 0.f;
    for (int j2 = 0; j2 < 64; ++j2) {
      const float e = __expf(row[j2] - mx);
      row[j2] = e;
      sum += e;
    }
    const float inv = 1.f / sum;
    // bf16 attn aliased over the head of this row's f32 storage. volatile keeps
    // the in-place f32->bf16 conversion ordered (reads stay ahead of writes).
    volatile float*          vrow = row;
    volatile unsigned short* arow = (volatile unsigned short*)row;
    for (int j2 = 0; j2 < 64; ++j2) {
      const float e = vrow[j2];
      arow[j2] = f2bf(e * inv);
    }
  }
  __syncthreads();

  // ============ Phase 4: O_h = V_h @ attn_h^T (16 tiles, 2/wave) ==============
#pragma unroll
  for (int j = 0; j < 2; ++j) {
    const int tile = wv * 2 + j;                       // 0..15
    const int h  = tile >> 2;
    const int n0 = (tile & 3) << 4;
    const unsigned short* attnT = (const unsigned short*)(S + h * (64 * LDSROW));
    v8f acc = {};
#pragma unroll
    for (int kb = 0; kb < 64; kb += 32) {
      v16bf a  = loadA32(Vd, h * HD, kb, LDB, lane);           // [d][j]
      v16bf bb = loadB32(attnT, n0, kb, 2 * LDSROW, lane);     // BT[i][j]
      acc = WMMA_BF16(a, bb, acc);
    }
    unsigned short* o = Ot + (n0 + N) * LDB + h * HD + mh;     // [token][chan]
    *(uint2*)(o)     = make_uint2(pk2(acc[0], acc[1]), pk2(acc[2], acc[3]));
    *(uint2*)(o + 4) = make_uint2(pk2(acc[4], acc[5]), pk2(acc[6], acc[7]));
  }
  __syncthreads();

  // ========= Phase 5: out = Wp @ O + bp, scatter to global (2 tiles/wave) =====
#pragma unroll
  for (int j = 0; j < 2; ++j) {
    const int tile = wv * 2 + j;
    const int m0 = (tile >> 2) << 4;
    const int n0 = (tile & 3) << 4;
    v8f acc = {};
#pragma unroll
    for (int kb = 0; kb < 64; kb += 32) {
      v16bf a  = loadA32(Wbf + 3 * (64 * LDB), m0, kb, LDB, lane);
      v16bf bb = loadB32(Ot, n0, kb, LDB, lane);
      acc = WMMA_BF16(a, bb, acc);
    }
    const float4 b0 = *(const float4*)(bias + 192 + m0 + mh);
    const float4 b1 = *(const float4*)(bias + 192 + m0 + mh + 4);
    const int l  = n0 + N;
    const int dz = l >> 4, hy = (l >> 2) & 3, wx = l & 3;
    const size_t sp = (size_t)(d0 + dz) * 4096 + (h0 + hy) * 64 + (w0 + wx);
    float* op = out + (size_t)(b * 64 + m0 + mh) * DHW + sp;
    op[0 * DHW] = acc[0] + b0.x;
    op[1 * DHW] = acc[1] + b0.y;
    op[2 * DHW] = acc[2] + b0.z;
    op[3 * DHW] = acc[3] + b0.w;
    op[4 * DHW] = acc[4] + b1.x;
    op[5 * DHW] = acc[5] + b1.y;
    op[6 * DHW] = acc[6] + b1.z;
    op[7 * DHW] = acc[7] + b1.w;
  }
}

extern "C" void kernel_launch(void* const* d_in, const int* in_sizes, int n_in,
                              void* d_out, int out_size, void* d_ws, size_t ws_size,
                              hipStream_t stream) {
  (void)n_in; (void)out_size; (void)d_ws; (void)ws_size;
  const float* x  = (const float*)d_in[0];
  const float* Wq = (const float*)d_in[1];
  const float* bq = (const float*)d_in[2];
  const float* Wk = (const float*)d_in[3];
  const float* bk = (const float*)d_in[4];
  const float* Wv = (const float*)d_in[5];
  const float* bv = (const float*)d_in[6];
  const float* Wp = (const float*)d_in[7];
  const float* bp = (const float*)d_in[8];

  const int B = in_sizes[0] / (C_DIM * DHW);           // batches (2)
  const int nWin = B * 16 * 16 * 16;                   // 8192 windows
  win_attn3d_kernel<<<nWin, 256, 0, stream>>>(x, Wq, bq, Wk, bk, Wv, bv, Wp, bp,
                                              (float*)d_out);
}
